// ATTLayer_19396072308956
// MI455X (gfx1250) — matrix operand order repaired and checked by reading, not compiled, not run on hardware
//
#include <hip/hip_runtime.h>
#include <hip/hip_bf16.h>
#include <math.h>

#define HID        128
#define HEADS      4
#define CCH        128
#define QKVS_COLS  1664   // 512(q) + 512(k) + 512(v) + 128(skip)

typedef __bf16 bf16_t;
typedef bf16_t v16bf __attribute__((ext_vector_type(16)));
typedef bf16_t v8bf  __attribute__((ext_vector_type(8)));
typedef float  v8f   __attribute__((ext_vector_type(8)));

// ---------------------------------------------------------------------------
// WMMA fragment loaders (contiguous k order chosen by this code, same for A and B, wave32)
// A (16x32 bf16): lanes 0-15 = row M=lane, K = 0..15;
//                 lanes 16-31 = row M=lane-16, K = 16..31
// ---------------------------------------------------------------------------
__device__ __forceinline__ v16bf load_a_frag(const bf16_t* __restrict__ A,
                                             int row, int k0, int lane) {
  const int kofs = (lane < 16) ? 0 : 16;
  const bf16_t* p = A + (size_t)row * HID + k0 + kofs;
  v8bf lo = *(const v8bf*)(p);        // K = k0+kofs .. +7   (16B)
  v8bf hi = *(const v8bf*)(p + 8);    // K = k0+kofs+8 .. +15
  v16bf a;
#pragma unroll
  for (int i = 0; i < 8; ++i) { a[i] = lo[i]; a[i + 8] = hi[i]; }
  return a;
}

// B (32x16 bf16), weights pre-transposed to WT[col][k] so each lane loads
// one contiguous run of 16 bf16: lanes 0-15 hold K=k0..k0+15 of col=lane,
// lanes 16-31 hold K=k0+16..k0+31 of col=lane-16.
__device__ __forceinline__ v16bf load_b_frag(const bf16_t* __restrict__ WT,
                                             int col, int k0, int lane) {
  const int kb = k0 + ((lane < 16) ? 0 : 16);
  return *(const v16bf*)(WT + (size_t)col * HID + kb);
}

// ---------------------------------------------------------------------------
// GEMM: C[r, c] = A[r, :128] x WT[c, :128]^T  (+ bias[c], optional ReLU)
// one wave per block (blockDim=32); each wave -> 16 rows x 64 cols
// ---------------------------------------------------------------------------
__global__ void k_wmma_gemm(const bf16_t* __restrict__ A,
                            const bf16_t* __restrict__ WT,
                            const float*  __restrict__ bias,
                            float* __restrict__ Cout,
                            int Nrows, int Ncols, int do_relu) {
  const int lane = threadIdx.x;
  const int r0   = blockIdx.x * 16;
  const int n0   = blockIdx.y * 64;
  const int arow = min(r0 + (lane & 15), Nrows - 1);
  const int coln = lane & 15;

  v8f acc[4] = {v8f{0.f}, v8f{0.f}, v8f{0.f}, v8f{0.f}};

#pragma unroll
  for (int kt = 0; kt < 4; ++kt) {               // K = 128 in steps of 32
    const int k0 = kt * 32;
    const v16bf a = load_a_frag(A, arow, k0, lane);
#pragma unroll
    for (int j = 0; j < 4; ++j) {
      const v16bf b = load_b_frag(WT, n0 + j * 16 + coln, k0, lane);
      acc[j] = __builtin_amdgcn_wmma_f32_16x16x32_bf16(
          /*neg_a=*/false, a, /*neg_b=*/false, b,
          /*c_mod=*/(short)0, acc[j], /*reuse_a=*/false, /*reuse_b=*/false);
    }
  }

  // C/D layout: VGPR v -> row r0 + v (+8 for lanes 16-31), col = n0+j*16+coln
  const int rbase = r0 + ((lane < 16) ? 0 : 8);
#pragma unroll
  for (int j = 0; j < 4; ++j) {
    const int col = n0 + j * 16 + coln;
    const float bv = bias[col];
#pragma unroll
    for (int v = 0; v < 8; ++v) {
      const int row = rbase + v;
      if (row < Nrows) {
        float x = acc[j][v] + bv;
        if (do_relu) x = fmaxf(x, 0.0f);
        Cout[(size_t)row * Ncols + col] = x;
      }
    }
  }
}

// ---------------------------------------------------------------------------
// Prep kernels
// ---------------------------------------------------------------------------
__global__ void k_cast_hx(const float* __restrict__ hx, bf16_t* __restrict__ hxb, int n) {
  int i = blockIdx.x * blockDim.x + threadIdx.x;
  if (i < n) hxb[i] = (bf16_t)hx[i];
}

__global__ void k_pack_w(const float* __restrict__ Wq, const float* __restrict__ Wk,
                         const float* __restrict__ Wv, const float* __restrict__ Ws,
                         bf16_t* __restrict__ wt) {
  int i = blockIdx.x * blockDim.x + threadIdx.x;     // QKVS_COLS*HID
  if (i >= QKVS_COLS * HID) return;
  const int col = i >> 7, kk = i & 127;
  float w;
  if      (col < 512)  w = Wq[(size_t)kk * 512 + col];
  else if (col < 1024) w = Wk[(size_t)kk * 512 + (col - 512)];
  else if (col < 1536) w = Wv[(size_t)kk * 512 + (col - 1024)];
  else                 w = Ws[(size_t)kk * 128 + (col - 1536)];
  wt[(size_t)col * HID + kk] = (bf16_t)w;
}

__global__ void k_pack_bias(const float* __restrict__ bq, const float* __restrict__ bk,
                            const float* __restrict__ bv, const float* __restrict__ bs,
                            float* __restrict__ ball) {
  int c = blockIdx.x * blockDim.x + threadIdx.x;
  if (c >= QKVS_COLS) return;
  float b;
  if      (c < 512)  b = bq[c];
  else if (c < 1024) b = bk[c - 512];
  else if (c < 1536) b = bv[c - 1024];
  else               b = bs[c - 1536];
  ball[c] = b;
}

__global__ void k_pack_up(const float* __restrict__ Wup, bf16_t* __restrict__ wupT) {
  int i = blockIdx.x * blockDim.x + threadIdx.x;     // 128*128
  if (i >= HID * HID) return;
  const int c = i >> 7, k = i & 127;
  wupT[(size_t)c * HID + k] = (bf16_t)Wup[(size_t)k * HID + c];
}

// c_t[c] = bup[c] + sum_j h_t[j] * Wup[128 + j][c]   (h_t half of concat is const)
__global__ void k_ct(const float* __restrict__ ht, const float* __restrict__ Wup,
                     const float* __restrict__ bup, float* __restrict__ ct) {
  int c = threadIdx.x;
  if (c >= HID) return;
  float s = bup[c];
  for (int j = 0; j < HID; ++j) s += ht[j] * Wup[(size_t)(HID + j) * HID + c];
  ct[c] = s;
}

__global__ void k_fill_neg_inf(float* __restrict__ p, int n) {
  int i = blockIdx.x * blockDim.x + threadIdx.x;
  if (i < n) p[i] = -INFINITY;
}

// ---------------------------------------------------------------------------
// Edge kernels
// ---------------------------------------------------------------------------
__device__ __forceinline__ void atomicMaxFloat(float* addr, float val) {
  // ordered-int trick (init must be -inf = 0xFF800000, valid for both paths)
  if (val >= 0.0f) atomicMax((int*)addr, __float_as_int(val));
  else             atomicMin((unsigned int*)addr, (unsigned int)__float_as_int(val));
}

// wave per edge: score[e,h] = dot(q[dst,h,:], k[src,h,:]) / sqrt(128)
__global__ void k_score(const int* __restrict__ ei, const float* __restrict__ qkvs,
                        float* __restrict__ score, float* __restrict__ m, int E) {
  const int gid = blockIdx.x * blockDim.x + threadIdx.x;
  const int e = gid >> 5;
  if (e >= E) return;
  const int lane = gid & 31;
  const int src = ei[e];
  const int dst = ei[E + e];
  const float4* q = (const float4*)(qkvs + (size_t)dst * QKVS_COLS) + lane * 4;
  const float4* k = (const float4*)(qkvs + (size_t)src * QKVS_COLS + 512) + lane * 4;
  float p = 0.f;
#pragma unroll
  for (int i = 0; i < 4; ++i) {
    const float4 qa = q[i]; const float4 ka = k[i];
    p += qa.x * ka.x + qa.y * ka.y + qa.z * ka.z + qa.w * ka.w;
  }
#pragma unroll
  for (int o = 1; o < 8; o <<= 1) p += __shfl_xor(p, o, 32);  // reduce head group
  if ((lane & 7) == 0) {
    const int h = lane >> 3;
    const float s = p * 0.08838834764831845f;  // 1/sqrt(128)
    score[(size_t)e * HEADS + h] = s;
    atomicMaxFloat(&m[(size_t)dst * HEADS + h], s);
  }
}

__global__ void k_exp(const int* __restrict__ ei, const float* __restrict__ score,
                      const float* __restrict__ m, float* __restrict__ ex,
                      float* __restrict__ denom, int E) {
  int idx = blockIdx.x * blockDim.x + threadIdx.x;
  if (idx >= E * HEADS) return;
  const int e = idx >> 2, h = idx & 3;
  const int dst = ei[E + e];
  const float x = __expf(score[idx] - m[(size_t)dst * HEADS + h]);
  ex[idx] = x;
  atomicAdd(&denom[(size_t)dst * HEADS + h], x);
}

// wave per edge: agg[dst,h,c] += alpha[e,h] * v[src,h,c]
__global__ void k_agg(const int* __restrict__ ei, const float* __restrict__ qkvs,
                      const float* __restrict__ ex, const float* __restrict__ denom,
                      float* __restrict__ agg, int E) {
  const int gid = blockIdx.x * blockDim.x + threadIdx.x;
  const int e = gid >> 5;
  if (e >= E) return;
  const int lane = gid & 31;
  const int src = ei[e];
  const int dst = ei[E + e];
  const int h = lane >> 3;
  const float alpha = ex[(size_t)e * HEADS + h] / denom[(size_t)dst * HEADS + h];
  const float4* v = (const float4*)(qkvs + (size_t)src * QKVS_COLS + 1024) + lane * 4;
  float* a = agg + (size_t)dst * (HEADS * CCH) + lane * 16;
#pragma unroll
  for (int i = 0; i < 4; ++i) {
    const float4 vv = v[i];
    atomicAdd(a + i * 4 + 0, alpha * vv.x);
    atomicAdd(a + i * 4 + 1, alpha * vv.y);
    atomicAdd(a + i * 4 + 2, alpha * vv.z);
    atomicAdd(a + i * 4 + 3, alpha * vv.w);
  }
}

// att[n,c] = mean_h agg[n,h,c] + skip[n,c]; cast bf16 for the update GEMM
__global__ void k_attout(const float* __restrict__ agg, const float* __restrict__ qkvs,
                         bf16_t* __restrict__ attb, int N) {
  int idx = blockIdx.x * blockDim.x + threadIdx.x;
  if (idx >= N * HID) return;
  const int n = idx >> 7, c = idx & 127;
  const float* ag = agg + (size_t)n * (HEADS * CCH);
  const float x = 0.25f * (ag[c] + ag[c + 128] + ag[c + 256] + ag[c + 384])
                + qkvs[(size_t)n * QKVS_COLS + 1536 + c];
  attb[idx] = (bf16_t)x;
}

// wave per row LayerNorm (128 elems, 4/lane)
__global__ void k_ln(const float* __restrict__ h, const float* __restrict__ gamma,
                     const float* __restrict__ beta, float* __restrict__ out, int N) {
  const int gid = blockIdx.x * blockDim.x + threadIdx.x;
  const int n = gid >> 5;
  if (n >= N) return;
  const int lane = gid & 31;
  const float4 x = *(const float4*)(h + (size_t)n * HID + lane * 4);
  float s = x.x + x.y + x.z + x.w;
#pragma unroll
  for (int o = 1; o < 32; o <<= 1) s += __shfl_xor(s, o, 32);
  const float mu = s * (1.0f / 128.0f);
  const float d0 = x.x - mu, d1 = x.y - mu, d2 = x.z - mu, d3 = x.w - mu;
  float v = d0 * d0 + d1 * d1 + d2 * d2 + d3 * d3;
#pragma unroll
  for (int o = 1; o < 32; o <<= 1) v += __shfl_xor(v, o, 32);
  const float rs = rsqrtf(v * (1.0f / 128.0f) + 1e-5f);
  const float4 g = *(const float4*)(gamma + lane * 4);
  const float4 b = *(const float4*)(beta + lane * 4);
  float4 o4;
  o4.x = d0 * rs * g.x + b.x;
  o4.y = d1 * rs * g.y + b.y;
  o4.z = d2 * rs * g.z + b.z;
  o4.w = d3 * rs * g.w + b.w;
  *(float4*)(out + (size_t)n * HID + lane * 4) = o4;
}

// ---------------------------------------------------------------------------
// Launch
// ---------------------------------------------------------------------------
static inline char* carve(char*& p, size_t bytes) {
  char* r = p;
  p += (bytes + 255) & ~(size_t)255;
  return r;
}

extern "C" void kernel_launch(void* const* d_in, const int* in_sizes, int n_in,
                              void* d_out, int out_size, void* d_ws, size_t ws_size,
                              hipStream_t stream) {
  const int*   ei    = (const int*)  d_in[0];
  const float* hX    = (const float*)d_in[1];
  const float* ht    = (const float*)d_in[2];
  const float* Wq    = (const float*)d_in[3];
  const float* bq    = (const float*)d_in[4];
  const float* Wk    = (const float*)d_in[5];
  const float* bk    = (const float*)d_in[6];
  const float* Wv    = (const float*)d_in[7];
  const float* bv    = (const float*)d_in[8];
  const float* Wskip = (const float*)d_in[9];
  const float* bskip = (const float*)d_in[10];
  const float* Wup   = (const float*)d_in[11];
  const float* bup   = (const float*)d_in[12];
  const float* gamma = (const float*)d_in[13];
  const float* beta  = (const float*)d_in[14];
  float* out = (float*)d_out;

  const int E = in_sizes[0] / 2;
  const int N = in_sizes[1] / HID;

  char* p = (char*)d_ws;
  bf16_t* hxb   = (bf16_t*)carve(p, (size_t)N * HID * sizeof(bf16_t));
  bf16_t* wtAll = (bf16_t*)carve(p, (size_t)QKVS_COLS * HID * sizeof(bf16_t));
  float*  bAll  = (float*) carve(p, (size_t)QKVS_COLS * sizeof(float));
  bf16_t* wupT  = (bf16_t*)carve(p, (size_t)HID * HID * sizeof(bf16_t));
  float*  ct    = (float*) carve(p, (size_t)HID * sizeof(float));
  float*  qkvs  = (float*) carve(p, (size_t)N * QKVS_COLS * sizeof(float));
  float*  score = (float*) carve(p, (size_t)E * HEADS * sizeof(float));
  float*  ex    = (float*) carve(p, (size_t)E * HEADS * sizeof(float));
  float*  mbuf  = (float*) carve(p, (size_t)N * HEADS * sizeof(float));
  float*  denom = (float*) carve(p, (size_t)N * HEADS * sizeof(float));
  float*  agg   = (float*) carve(p, (size_t)N * HEADS * CCH * sizeof(float));
  bf16_t* attb  = (bf16_t*)carve(p, (size_t)N * HID * sizeof(bf16_t));
  float*  hbuf  = (float*) carve(p, (size_t)N * HID * sizeof(float));

  const int TB = 256;

  // ---- prep ----
  hipMemsetAsync(denom, 0, (size_t)N * HEADS * sizeof(float), stream);
  hipMemsetAsync(agg,   0, (size_t)N * HEADS * CCH * sizeof(float), stream);
  k_fill_neg_inf<<<(N * HEADS + TB - 1) / TB, TB, 0, stream>>>(mbuf, N * HEADS);
  k_cast_hx<<<((size_t)N * HID + TB - 1) / TB, TB, 0, stream>>>(hX, hxb, N * HID);
  k_pack_w<<<(QKVS_COLS * HID + TB - 1) / TB, TB, 0, stream>>>(Wq, Wk, Wv, Wskip, wtAll);
  k_pack_bias<<<(QKVS_COLS + TB - 1) / TB, TB, 0, stream>>>(bq, bk, bv, bskip, bAll);
  k_pack_up<<<(HID * HID + TB - 1) / TB, TB, 0, stream>>>(Wup, wupT);
  k_ct<<<1, HID, 0, stream>>>(ht, Wup, bup, ct);

  // ---- fused Q|K|V|skip projection: [N x 128] x [128 x 1664] ----
  {
    dim3 grid((N + 15) / 16, QKVS_COLS / 64);
    k_wmma_gemm<<<grid, 32, 0, stream>>>(hxb, wtAll, bAll, qkvs, N, QKVS_COLS, 0);
  }

  // ---- edge attention ----
  {
    const long long t = (long long)E * 32;
    k_score<<<(unsigned)((t + TB - 1) / TB), TB, 0, stream>>>(ei, qkvs, score, mbuf, E);
  }
  k_exp<<<(E * HEADS + TB - 1) / TB, TB, 0, stream>>>(ei, score, mbuf, ex, denom, E);
  {
    const long long t = (long long)E * 32;
    k_agg<<<(unsigned)((t + TB - 1) / TB), TB, 0, stream>>>(ei, qkvs, ex, denom, agg, E);
  }
  k_attout<<<((size_t)N * HID + TB - 1) / TB, TB, 0, stream>>>(agg, qkvs, attb, N);

  // ---- update GEMM (+c_t bias, ReLU), then LayerNorm ----
  {
    dim3 grid((N + 15) / 16, HID / 64);
    k_wmma_gemm<<<grid, 32, 0, stream>>>(attb, wupT, ct, hbuf, N, HID, 1);
  }
  {
    const long long t = (long long)N * 32;
    k_ln<<<(unsigned)((t + TB - 1) / TB), TB, 0, stream>>>(hbuf, gamma, beta, out, N);
  }
}
